// MultiHeadAttention_28166395527549
// MI455X (gfx1250) — compile-verified, matrix operand order checked by reading, not run on hardware
//
#include <hip/hip_runtime.h>

#define B_   4
#define N_   2048
#define EMB_ 768
#define H_   12
#define D_   64
#define M_   (B_ * N_)     // 8192 rows
#define K3_  (3 * EMB_)    // 2304

typedef __attribute__((ext_vector_type(16))) __bf16       v16bf;
typedef __attribute__((ext_vector_type(8)))  float        v8f;
typedef __attribute__((ext_vector_type(4)))  unsigned int u32x4;

union Frag16 { v16bf v; u32x4 u[2]; };

__device__ __forceinline__ unsigned short f2bf(float f) {
  unsigned int u = __float_as_uint(f);
  u += 0x7FFFu + ((u >> 16) & 1u);   // round-to-nearest-even
  return (unsigned short)(u >> 16);
}

// Load a 16x32 bf16 A/B fragment from a K-major operand:
// element (r, k) lives at base[r*stride + k], r in [0,16), k in [0,32).
// CDNA5 16-bit A layout: lane L holds row L&15; k-halves {kb..kb+7, kb+16..kb+23},
// kb = 8*(L>>4). Two 16-byte loads per lane.
__device__ __forceinline__ v16bf load_ab(const unsigned short* base, int stride, int lane) {
  int r  = lane & 15;
  int kb = (lane >> 4) << 3;
  const unsigned short* p = base + (size_t)r * stride + kb;
  Frag16 f;
  f.u[0] = *(const u32x4*)(p);
  f.u[1] = *(const u32x4*)(p + 16);
  return f.v;
}

__global__ void cvt_f32_bf16(const float* __restrict__ src,
                             unsigned short* __restrict__ dst, int n) {
  int i = blockIdx.x * blockDim.x + threadIdx.x;
  int stride = gridDim.x * blockDim.x;
  for (; i < n; i += stride) dst[i] = f2bf(src[i]);
}

// ---------------- QKV GEMM: [8192,768] x [2304,768]^T -> Q,K (b,h,n,d) and V^T (b,h,d,n)
__global__ void __launch_bounds__(256)
gemm_qkv(const unsigned short* __restrict__ A,   // x  bf16 [8192,768]
         const unsigned short* __restrict__ W,   // Wqkv bf16 [2304,768]
         unsigned short* __restrict__ Qo,        // [48,2048,64]
         unsigned short* __restrict__ Ko,        // [48,2048,64]
         unsigned short* __restrict__ Vt)        // [48,64,2048]
{
  int lane = threadIdx.x & 31;
  int tid  = blockIdx.x * 8 + (threadIdx.x >> 5);   // 18432 waves
  int j0 = (tid % 72) * 32;                          // col tile (2304/32)
  int m0 = (tid / 72) * 32;                          // row tile (8192/32)

  v8f c00 = {}, c01 = {}, c10 = {}, c11 = {};
  for (int k0 = 0; k0 < 768; k0 += 32) {
    v16bf a0 = load_ab(A + (size_t)m0 * 768 + k0,        768, lane);
    v16bf a1 = load_ab(A + (size_t)(m0 + 16) * 768 + k0, 768, lane);
    v16bf b0 = load_ab(W + (size_t)j0 * 768 + k0,        768, lane);
    v16bf b1 = load_ab(W + (size_t)(j0 + 16) * 768 + k0, 768, lane);
    c00 = __builtin_amdgcn_wmma_f32_16x16x32_bf16(false, a0, false, b0, (short)0, c00, false, false);
    c01 = __builtin_amdgcn_wmma_f32_16x16x32_bf16(false, a0, false, b1, (short)0, c01, false, false);
    c10 = __builtin_amdgcn_wmma_f32_16x16x32_bf16(false, a1, false, b0, (short)0, c10, false, false);
    c11 = __builtin_amdgcn_wmma_f32_16x16x32_bf16(false, a1, false, b1, (short)0, c11, false, false);
  }

  int b   = m0 / N_;
  int t0  = m0 % N_;
  int sel = j0 / EMB_;             // 0=Q 1=K 2=V
  int wi  = j0 % EMB_;
  int h   = wi / D_;
  int d0  = wi % D_;
  int bh  = b * H_ + h;
  int n   = lane & 15;
  int mo  = (lane >> 4) << 3;

  v8f cf[2][2] = { { c00, c01 }, { c10, c11 } };
  if (sel < 2) {
    unsigned short* dst = (sel == 0 ? Qo : Ko) + (size_t)bh * N_ * D_;
    for (int i = 0; i < 2; ++i)
      for (int jj = 0; jj < 2; ++jj) {
        unsigned short* p = dst + (size_t)(t0 + 16 * i) * D_ + d0 + 16 * jj;
        for (int r = 0; r < 8; ++r)
          p[(size_t)(mo + r) * D_ + n] = f2bf(cf[i][jj][r]);
      }
  } else {
    // transposed store: V^T[d][token]
    unsigned short* dst = Vt + (size_t)bh * D_ * N_;
    for (int i = 0; i < 2; ++i)
      for (int jj = 0; jj < 2; ++jj) {
        unsigned short* p = dst + (size_t)(d0 + 16 * jj) * N_ + t0 + 16 * i;
        for (int r = 0; r < 8; ++r)
          p[(size_t)n * N_ + mo + r] = f2bf(cf[i][jj][r]);
      }
  }
}

// ---------------- Flash attention: one wave = 16 query rows of one (b,h)
__global__ void __launch_bounds__(256)
attn_fwd(const unsigned short* __restrict__ Q,    // [48,2048,64]
         const unsigned short* __restrict__ K,    // [48,2048,64]
         const unsigned short* __restrict__ Vt,   // [48,64,2048]
         unsigned short* __restrict__ O)          // [8192,768] bf16
{
  __shared__ unsigned short plds[8 * 16 * 32];   // 1KB per wave P staging
  int lane  = threadIdx.x & 31;
  int wslot = threadIdx.x >> 5;
  int tid   = blockIdx.x * 8 + wslot;            // 6144 waves
  int bh = tid >> 7;                              // /128 q-tiles
  int q0 = (tid & 127) << 4;

  const unsigned short* Qp = Q  + (size_t)bh * N_ * D_;
  const unsigned short* Kp = K  + (size_t)bh * N_ * D_;
  const unsigned short* Vp = Vt + (size_t)bh * D_ * N_;

  v16bf qf0 = load_ab(Qp + (size_t)q0 * D_ + 0,  D_, lane);
  v16bf qf1 = load_ab(Qp + (size_t)q0 * D_ + 32, D_, lane);

  v8f o0 = {}, o1 = {}, o2 = {}, o3 = {};
  float mrun[8], lrun[8];
  for (int r = 0; r < 8; ++r) { mrun[r] = -3.0e38f; lrun[r] = 0.0f; }

  unsigned short* lp = plds + wslot * 512;
  int n    = lane & 15;
  int half = lane >> 4;

  for (int kb0 = 0; kb0 < N_; kb0 += 32) {
    v16bf k00 = load_ab(Kp + (size_t)kb0 * D_ + 0,         D_, lane);
    v16bf k01 = load_ab(Kp + (size_t)kb0 * D_ + 32,        D_, lane);
    v16bf k10 = load_ab(Kp + (size_t)(kb0 + 16) * D_ + 0,  D_, lane);
    v16bf k11 = load_ab(Kp + (size_t)(kb0 + 16) * D_ + 32, D_, lane);

    v8f s0 = {}, s1 = {};
    s0 = __builtin_amdgcn_wmma_f32_16x16x32_bf16(false, qf0, false, k00, (short)0, s0, false, false);
    s0 = __builtin_amdgcn_wmma_f32_16x16x32_bf16(false, qf1, false, k01, (short)0, s0, false, false);
    s1 = __builtin_amdgcn_wmma_f32_16x16x32_bf16(false, qf0, false, k10, (short)0, s1, false, false);
    s1 = __builtin_amdgcn_wmma_f32_16x16x32_bf16(false, qf1, false, k11, (short)0, s1, false, false);

    // online softmax; masks 1,2,4,8 reduce within each 16-lane half (rows r and r+8)
    for (int r = 0; r < 8; ++r) {
      float a = s0[r] * 0.125f;      // SCALE = 1/sqrt(64)
      float c = s1[r] * 0.125f;
      float rm = fmaxf(a, c);
      rm = fmaxf(rm, __shfl_xor(rm, 1, 32));
      rm = fmaxf(rm, __shfl_xor(rm, 2, 32));
      rm = fmaxf(rm, __shfl_xor(rm, 4, 32));
      rm = fmaxf(rm, __shfl_xor(rm, 8, 32));
      float nm    = fmaxf(mrun[r], rm);
      float alpha = __expf(mrun[r] - nm);
      float p0 = __expf(a - nm);
      float p1 = __expf(c - nm);
      float rs = p0 + p1;
      rs += __shfl_xor(rs, 1, 32);
      rs += __shfl_xor(rs, 2, 32);
      rs += __shfl_xor(rs, 4, 32);
      rs += __shfl_xor(rs, 8, 32);
      lrun[r] = lrun[r] * alpha + rs;
      mrun[r] = nm;
      o0[r] *= alpha; o1[r] *= alpha; o2[r] *= alpha; o3[r] *= alpha;
      int row = r + 8 * half;
      lp[row * 32 + n]      = f2bf(p0);
      lp[row * 32 + 16 + n] = f2bf(p1);
    }

    v16bf pa = load_ab(lp, 32, lane);   // same-wave DS RAW: in-order per ISA
    v16bf v0 = load_ab(Vp + (size_t) 0 * N_ + kb0, N_, lane);
    v16bf v1 = load_ab(Vp + (size_t)16 * N_ + kb0, N_, lane);
    v16bf v2 = load_ab(Vp + (size_t)32 * N_ + kb0, N_, lane);
    v16bf v3 = load_ab(Vp + (size_t)48 * N_ + kb0, N_, lane);
    o0 = __builtin_amdgcn_wmma_f32_16x16x32_bf16(false, pa, false, v0, (short)0, o0, false, false);
    o1 = __builtin_amdgcn_wmma_f32_16x16x32_bf16(false, pa, false, v1, (short)0, o1, false, false);
    o2 = __builtin_amdgcn_wmma_f32_16x16x32_bf16(false, pa, false, v2, (short)0, o2, false, false);
    o3 = __builtin_amdgcn_wmma_f32_16x16x32_bf16(false, pa, false, v3, (short)0, o3, false, false);
  }

  for (int r = 0; r < 8; ++r) {
    float inv = 1.0f / lrun[r];
    o0[r] *= inv; o1[r] *= inv; o2[r] *= inv; o3[r] *= inv;
  }

  int b = bh / H_, h = bh % H_;
  unsigned short* op = O + (size_t)(b * N_ + q0) * EMB_ + h * D_;
  v8f of[4] = { o0, o1, o2, o3 };
  for (int g = 0; g < 4; ++g)
    for (int r = 0; r < 8; ++r)
      op[(size_t)(r + 8 * half) * EMB_ + g * 16 + n] = f2bf(of[g][r]);
}

// ---------------- Projection: [8192,768] x [768,768]^T + bias -> f32 out
__global__ void __launch_bounds__(256)
gemm_proj(const unsigned short* __restrict__ A,   // attn bf16 [8192,768]
          const unsigned short* __restrict__ W,   // Wproj bf16 [768,768]
          const float* __restrict__ bias,
          float* __restrict__ out)                // [8192,768] f32
{
  int lane = threadIdx.x & 31;
  int tid  = blockIdx.x * 8 + (threadIdx.x >> 5); // 6144 waves
  int j0 = (tid % 24) * 32;
  int m0 = (tid / 24) * 32;

  v8f c00 = {}, c01 = {}, c10 = {}, c11 = {};
  for (int k0 = 0; k0 < 768; k0 += 32) {
    v16bf a0 = load_ab(A + (size_t)m0 * 768 + k0,        768, lane);
    v16bf a1 = load_ab(A + (size_t)(m0 + 16) * 768 + k0, 768, lane);
    v16bf b0 = load_ab(W + (size_t)j0 * 768 + k0,        768, lane);
    v16bf b1 = load_ab(W + (size_t)(j0 + 16) * 768 + k0, 768, lane);
    c00 = __builtin_amdgcn_wmma_f32_16x16x32_bf16(false, a0, false, b0, (short)0, c00, false, false);
    c01 = __builtin_amdgcn_wmma_f32_16x16x32_bf16(false, a0, false, b1, (short)0, c01, false, false);
    c10 = __builtin_amdgcn_wmma_f32_16x16x32_bf16(false, a1, false, b0, (short)0, c10, false, false);
    c11 = __builtin_amdgcn_wmma_f32_16x16x32_bf16(false, a1, false, b1, (short)0, c11, false, false);
  }

  int n  = lane & 15;
  int mo = (lane >> 4) << 3;
  v8f cf[2][2] = { { c00, c01 }, { c10, c11 } };
  for (int i = 0; i < 2; ++i)
    for (int jj = 0; jj < 2; ++jj) {
      float bv = bias[j0 + 16 * jj + n];
      float* p = out + (size_t)(m0 + 16 * i) * 768 + j0 + 16 * jj;
      for (int r = 0; r < 8; ++r)
        p[(size_t)(mo + r) * 768 + n] = cf[i][jj][r] + bv;
    }
}

extern "C" void kernel_launch(void* const* d_in, const int* in_sizes, int n_in,
                              void* d_out, int out_size, void* d_ws, size_t ws_size,
                              hipStream_t stream) {
  const float* x     = (const float*)d_in[0];   // [4,2048,768]
  const float* Wqkv  = (const float*)d_in[1];   // [2304,768]
  const float* Wproj = (const float*)d_in[2];   // [768,768]
  const float* bproj = (const float*)d_in[3];   // [768]
  float* out = (float*)d_out;

  const size_t nX  = (size_t)M_ * EMB_;      // 6,291,456
  const size_t nWq = (size_t)K3_ * EMB_;     // 1,769,472
  const size_t nWp = (size_t)EMB_ * EMB_;    //   589,824
  const size_t nQ  = (size_t)B_ * H_ * N_ * D_; // 6,291,456 (each of Q,K,Vt)

  char* ws = (char*)d_ws;
  unsigned short* xb   = (unsigned short*)(ws);
  unsigned short* wqb  = xb  + nX;
  unsigned short* wpb  = wqb + nWq;
  unsigned short* Qb   = wpb + nWp;
  unsigned short* Kb   = Qb  + nQ;
  unsigned short* Vtb  = Kb  + nQ;
  unsigned short* Ab   = Vtb + nQ;           // attn out bf16 [8192,768]

  cvt_f32_bf16<<<1024, 256, 0, stream>>>(x,     xb,  (int)nX);
  cvt_f32_bf16<<<512,  256, 0, stream>>>(Wqkv,  wqb, (int)nWq);
  cvt_f32_bf16<<<256,  256, 0, stream>>>(Wproj, wpb, (int)nWp);

  gemm_qkv <<<2304, 256, 0, stream>>>(xb, wqb, Qb, Kb, Vtb);   // 18432 waves
  attn_fwd <<<768,  256, 0, stream>>>(Qb, Kb, Vtb, Ab);        //  6144 waves
  gemm_proj<<<768,  256, 0, stream>>>(Ab, wpb, bproj, out);    //  6144 waves
}
